// ArcticMoE_11699490914928
// MI455X (gfx1250) — matrix operand order repaired and checked by reading, not compiled, not run on hardware
//
#include <hip/hip_runtime.h>

#define T_DIM 4096
#define H_DIM 2048
#define I_DIM 2048
#define E_NUM 16
#define LDA 40   // padded LDS row stride in bf16 elems (80 B, conflict-avoiding)

typedef __bf16 v16bf __attribute__((ext_vector_type(16)));
typedef __bf16 v8bf  __attribute__((ext_vector_type(8)));
typedef __bf16 v4bf  __attribute__((ext_vector_type(4)));
typedef float  v8f   __attribute__((ext_vector_type(8)));

// ---- CDNA5 async global->LDS copy (16 B per lane), tracked by ASYNCcnt ----
__device__ __forceinline__ void async_copy16(unsigned lds_off, unsigned long long gaddr) {
  asm volatile("global_load_async_to_lds_b128 %0, %1, off"
               :: "v"(lds_off), "v"(gaddr) : "memory");
}
__device__ __forceinline__ void wait_async0() {
  asm volatile("s_wait_asynccnt 0x0" ::: "memory");
}
__device__ __forceinline__ unsigned long long gaddr_of(const void* p) {
  return (unsigned long long)(uintptr_t)p;
}
__device__ __forceinline__ unsigned lds_off_of(const void* p) {
  return (unsigned)(uintptr_t)p;   // low 32 bits of generic shared ptr = LDS byte offset
}
__device__ __forceinline__ v8bf cvt8(float4 a, float4 b) {
  v8bf v;
  v[0]=(__bf16)a.x; v[1]=(__bf16)a.y; v[2]=(__bf16)a.z; v[3]=(__bf16)a.w;
  v[4]=(__bf16)b.x; v[5]=(__bf16)b.y; v[6]=(__bf16)b.z; v[7]=(__bf16)b.w;
  return v;
}

// ---- fragment builders (wave32, 16x16x32 bf16 WMMA) ----
// A (16x32): lane L holds row L&15; elems 0..7 = K[(L>=16?8:0)+0..7], elems 8..15 = K[16+...]
__device__ __forceinline__ v16bf load_frag_a(const __bf16* sbase, int lane) {
  int r  = lane & 15;
  int k8 = (lane >> 4) << 3;
  const __bf16* row = sbase + r * LDA + k8;
  v8bf lo = *(const v8bf*)(row);
  v8bf hi = *(const v8bf*)(row + 16);
  return __builtin_shufflevector(lo, hi, 0,1,2,3,4,5,6,7,8,9,10,11,12,13,14,15);
}
// B (32x16): lane L holds col L&15; elems 0..7 = K[(L>=16?8:0)+0..7], elems 8..15 = K[16+...]
__device__ __forceinline__ v16bf load_frag_b(const __bf16* sbase, int lane) {
  int c   = lane & 15;
  int k16 = (lane >> 4) << 3;
  const __bf16* row = sbase + c * LDA + k16;
  v8bf lo = *(const v8bf*)(row);
  v8bf hi = *(const v8bf*)(row + 16);
  return __builtin_shufflevector(lo, hi, 0,1,2,3,4,5,6,7,8,9,10,11,12,13,14,15);
}

// ---- kernel 0: zero output + expert counters ----
__global__ __launch_bounds__(256) void moe_zero(float* __restrict__ out, int n4,
                                                int* __restrict__ cnt) {
  int i = blockIdx.x * 256 + threadIdx.x;
  if (i < n4) {
    float4 z = make_float4(0.f, 0.f, 0.f, 0.f);
    ((float4*)out)[i] = z;
  }
  if (i < E_NUM) cnt[i] = 0;
}

// ---- kernel 1: gating (logits -> softmax -> top2 -> renorm -> route) ----
__global__ __launch_bounds__(256) void moe_gate(const float* __restrict__ x,
                                                const float* __restrict__ gw,
                                                float* __restrict__ coefs,
                                                int* __restrict__ toklist,
                                                int* __restrict__ cnt) {
  const int t = blockIdx.x;
  const int tid = threadIdx.x, lane = tid & 31, wid = tid >> 5;
  __shared__ float slog[E_NUM];
  const float* xr = x + (size_t)t * H_DIM;
  for (int ep = 0; ep < 2; ++ep) {
    int e = wid * 2 + ep;
    const float* w = gw + (size_t)e * H_DIM;
    float s = 0.f;
    for (int k = lane; k < H_DIM; k += 32) s += xr[k] * w[k];
    for (int off = 16; off > 0; off >>= 1) s += __shfl_xor(s, off, 32);
    if (lane == 0) slog[e] = s;
  }
  __syncthreads();
  if (tid == 0) {
    float mx = slog[0];
    for (int e = 1; e < E_NUM; ++e) mx = fmaxf(mx, slog[e]);
    float p[E_NUM];
    for (int e = 0; e < E_NUM; ++e) p[e] = expf(slog[e] - mx);
    int i0 = 0; float v0 = p[0];
    for (int e = 1; e < E_NUM; ++e) if (p[e] > v0) { v0 = p[e]; i0 = e; }
    int i1 = -1; float v1 = -1.f;
    for (int e = 0; e < E_NUM; ++e) if (e != i0 && p[e] > v1) { v1 = p[e]; i1 = e; }
    float inv = 1.f / (v0 + v1);   // softmax denominator cancels in renorm
    coefs[t * 2 + 0] = v0 * inv;
    coefs[t * 2 + 1] = v1 * inv;
    int p0 = atomicAdd(&cnt[i0], 1); toklist[i0 * T_DIM + p0] = t * 2;
    int p1 = atomicAdd(&cnt[i1], 1); toklist[i1 * T_DIM + p1] = t * 2 + 1;
  }
}

// ---- kernel 2: x fp32 -> bf16 ----
__global__ __launch_bounds__(256) void moe_cvt(const float* __restrict__ x,
                                               __bf16* __restrict__ xb) {
  int i = blockIdx.x * 256 + threadIdx.x;
  float4 f = ((const float4*)x)[i];
  v4bf b;
  b[0] = (__bf16)f.x; b[1] = (__bf16)f.y; b[2] = (__bf16)f.z; b[3] = (__bf16)f.w;
  ((v4bf*)xb)[i] = b;
}

// ---- kernel 3: routed GEMM1  h = silu(x@Wg^T) * (x@Wu^T)  per expert ----
// 256 thr (8 waves); tile 128 rows x 64 I-cols; wave 32x32; 2x2 subtiles x {g,u}
// double-buffered LDS pipeline: async A copy + weight loads for stage k+1 overlap WMMA of stage k
__global__ __launch_bounds__(256) void moe_gemm1(const __bf16* __restrict__ xb,
                                                 const float* __restrict__ ws,
                                                 __bf16* __restrict__ h,
                                                 const int* __restrict__ toklist,
                                                 const int* __restrict__ cnt) {
  const int e = blockIdx.x;
  const int ne = cnt[e];
  const int m0 = blockIdx.y * 128;
  if (m0 >= ne) return;
  const int n0 = blockIdx.z * 64;

  __shared__ __attribute__((aligned(16))) __bf16 sA[2][128 * LDA];
  __shared__ __attribute__((aligned(16))) __bf16 sBg[2][64 * LDA];
  __shared__ __attribute__((aligned(16))) __bf16 sBu[2][64 * LDA];
  __shared__ int sRow[128];

  const int tid = threadIdx.x;
  const int lane = tid & 31;
  const int wid = tid >> 5;
  const int wm0 = (wid & 3) * 32;
  const int wn0 = (wid >> 2) * 32;

  const int* tl = toklist + e * T_DIM;
  {
    int idx = m0 + tid;
    if (tid < 128) sRow[tid] = (idx < ne) ? tl[idx] : -1;
  }
  __syncthreads();

  // per-thread staging slots
  const int ar  = tid >> 1;             // A: 0..127 rows, 32 B half-row each
  const int ack = (tid & 1) << 4;
  int arow = sRow[ar]; if (arow < 0) arow = 0;   // clamp; masked rows dropped in epilogue
  const __bf16* agsrc = xb + (size_t)(arow >> 1) * H_DIM + ack;
  const unsigned aldsA[2] = { lds_off_of(&sA[0][ar * LDA + ack]),
                              lds_off_of(&sA[1][ar * LDA + ack]) };

  const int br  = tid >> 2;             // B: 64 rows x 4 chunks of 8
  const int bck = (tid & 3) << 3;
  const int bofs = br * LDA + bck;      // LDS elem offset for this thread's B slot
  const float* wsg = ws + (size_t)e * (2 * I_DIM) * H_DIM + (size_t)n0 * H_DIM;
  const float* wsu = wsg + (size_t)I_DIM * H_DIM;
  const float* bsg = wsg + (size_t)br * H_DIM + bck;
  const float* bsu = wsu + (size_t)br * H_DIM + bck;

  v8f accg[2][2] = {};
  v8f accu[2][2] = {};

  // prologue: stage k0 = 0 into buffer 0
  async_copy16(aldsA[0],      gaddr_of(agsrc));
  async_copy16(aldsA[0] + 16, gaddr_of(agsrc + 8));
  {
    float4 g0 = *(const float4*)bsg, g1 = *(const float4*)(bsg + 4);
    float4 u0 = *(const float4*)bsu, u1 = *(const float4*)(bsu + 4);
    *(v8bf*)&sBg[0][bofs] = cvt8(g0, g1);   // direct shared indexing -> ds_store
    *(v8bf*)&sBu[0][bofs] = cvt8(u0, u1);
  }

  int buf = 0;
  for (int k0 = 0; k0 < H_DIM; k0 += 32) {
    wait_async0();
    __syncthreads();
    const int nb = buf ^ 1;
    const bool more = (k0 + 32) < H_DIM;
    float4 g0 = {}, g1 = {}, u0 = {}, u1 = {};
    if (more) {
      // issue next-stage A async copy + B global loads; cvt/store sunk below WMMAs
      async_copy16(aldsA[nb],      gaddr_of(agsrc + k0 + 32));
      async_copy16(aldsA[nb] + 16, gaddr_of(agsrc + k0 + 40));
      const float* pg = bsg + k0 + 32;
      const float* pu = bsu + k0 + 32;
      g0 = *(const float4*)pg; g1 = *(const float4*)(pg + 4);
      u0 = *(const float4*)pu; u1 = *(const float4*)(pu + 4);
      if (k0 + 64 < H_DIM) { __builtin_prefetch(pg + 32, 0, 1); __builtin_prefetch(pu + 32, 0, 1); }
    }

    v16bf a0  = load_frag_a(&sA[buf][(wm0 + 0)  * LDA], lane);
    v16bf a1  = load_frag_a(&sA[buf][(wm0 + 16) * LDA], lane);
    v16bf bg0 = load_frag_b(&sBg[buf][(wn0 + 0)  * LDA], lane);
    v16bf bg1 = load_frag_b(&sBg[buf][(wn0 + 16) * LDA], lane);
    v16bf bu0 = load_frag_b(&sBu[buf][(wn0 + 0)  * LDA], lane);
    v16bf bu1 = load_frag_b(&sBu[buf][(wn0 + 16) * LDA], lane);

    accg[0][0] = __builtin_amdgcn_wmma_f32_16x16x32_bf16(false, a0, false, bg0, (short)0, accg[0][0], false, false);
    accg[0][1] = __builtin_amdgcn_wmma_f32_16x16x32_bf16(false, a0, false, bg1, (short)0, accg[0][1], false, false);
    accg[1][0] = __builtin_amdgcn_wmma_f32_16x16x32_bf16(false, a1, false, bg0, (short)0, accg[1][0], false, false);
    accg[1][1] = __builtin_amdgcn_wmma_f32_16x16x32_bf16(false, a1, false, bg1, (short)0, accg[1][1], false, false);
    accu[0][0] = __builtin_amdgcn_wmma_f32_16x16x32_bf16(false, a0, false, bu0, (short)0, accu[0][0], false, false);
    accu[0][1] = __builtin_amdgcn_wmma_f32_16x16x32_bf16(false, a0, false, bu1, (short)0, accu[0][1], false, false);
    accu[1][0] = __builtin_amdgcn_wmma_f32_16x16x32_bf16(false, a1, false, bu0, (short)0, accu[1][0], false, false);
    accu[1][1] = __builtin_amdgcn_wmma_f32_16x16x32_bf16(false, a1, false, bu1, (short)0, accu[1][1], false, false);

    if (more) {
      *(v8bf*)&sBg[nb][bofs] = cvt8(g0, g1);
      *(v8bf*)&sBu[nb][bofs] = cvt8(u0, u1);
    }
    buf = nb;
  }

  // epilogue: silu(g)*u -> bf16 h (C layout: elem i -> row i+8*(lane>=16), col lane&15)
  const int mlane = (lane >> 4) << 3;
  const int ncol = lane & 15;
  for (int ms = 0; ms < 2; ++ms)
    for (int ns = 0; ns < 2; ++ns) {
      v8f g = accg[ms][ns], u = accu[ms][ns];
      for (int i = 0; i < 8; ++i) {
        int rloc = wm0 + ms * 16 + mlane + i;
        int rowid = sRow[rloc];
        if (rowid >= 0) {
          float gv = g[i];
          float hv = (gv / (1.f + expf(-gv))) * u[i];
          int j = n0 + wn0 + ns * 16 + ncol;
          h[(size_t)rowid * I_DIM + j] = (__bf16)hv;
        }
      }
    }
}

// ---- kernel 4: routed GEMM2  out[t] += coef * (h @ W2^T)  per expert ----
// block tile 128x128; wave 32x64 -> 2x4 subtiles; same double-buffered pipeline
__global__ __launch_bounds__(256) void moe_gemm2(const __bf16* __restrict__ h,
                                                 const float* __restrict__ w2s,
                                                 float* __restrict__ out,
                                                 const int* __restrict__ toklist,
                                                 const float* __restrict__ coefs,
                                                 const int* __restrict__ cnt) {
  const int e = blockIdx.x;
  const int ne = cnt[e];
  const int m0 = blockIdx.y * 128;
  if (m0 >= ne) return;
  const int n0 = blockIdx.z * 128;

  __shared__ __attribute__((aligned(16))) __bf16 sA[2][128 * LDA];
  __shared__ __attribute__((aligned(16))) __bf16 sB[2][128 * LDA];
  __shared__ int sRow[128];

  const int tid = threadIdx.x;
  const int lane = tid & 31;
  const int wid = tid >> 5;
  const int wm0 = (wid & 3) * 32;
  const int wn0 = (wid >> 2) * 64;

  const int* tl = toklist + e * T_DIM;
  {
    int idx = m0 + tid;
    if (tid < 128) sRow[tid] = (idx < ne) ? tl[idx] : -1;
  }
  __syncthreads();

  const int ar  = tid >> 1;
  const int ack = (tid & 1) << 4;
  int arow = sRow[ar]; if (arow < 0) arow = 0;   // h rows 0..2T-1 all written by gemm1
  const __bf16* agsrc = h + (size_t)arow * I_DIM + ack;
  const unsigned aldsA[2] = { lds_off_of(&sA[0][ar * LDA + ack]),
                              lds_off_of(&sA[1][ar * LDA + ack]) };

  const int br  = tid >> 2;             // B: rows br and br+64, 4 chunks of 8
  const int bck = (tid & 3) << 3;
  const int bofs0 = br * LDA + bck;
  const int bofs1 = (br + 64) * LDA + bck;
  const float* w2 = w2s + (size_t)e * H_DIM * I_DIM + (size_t)n0 * I_DIM;
  const float* bs0 = w2 + (size_t)br * I_DIM + bck;
  const float* bs1 = w2 + (size_t)(br + 64) * I_DIM + bck;

  v8f acc[2][4] = {};

  // prologue: stage k0 = 0 into buffer 0
  async_copy16(aldsA[0],      gaddr_of(agsrc));
  async_copy16(aldsA[0] + 16, gaddr_of(agsrc + 8));
  {
    float4 a0 = *(const float4*)bs0, a1 = *(const float4*)(bs0 + 4);
    float4 b0 = *(const float4*)bs1, b1 = *(const float4*)(bs1 + 4);
    *(v8bf*)&sB[0][bofs0] = cvt8(a0, a1);
    *(v8bf*)&sB[0][bofs1] = cvt8(b0, b1);
  }

  int buf = 0;
  for (int k0 = 0; k0 < I_DIM; k0 += 32) {
    wait_async0();
    __syncthreads();
    const int nb = buf ^ 1;
    const bool more = (k0 + 32) < I_DIM;
    float4 f0 = {}, f1 = {}, f2 = {}, f3 = {};
    if (more) {
      async_copy16(aldsA[nb],      gaddr_of(agsrc + k0 + 32));
      async_copy16(aldsA[nb] + 16, gaddr_of(agsrc + k0 + 40));
      const float* p0 = bs0 + k0 + 32;
      const float* p1 = bs1 + k0 + 32;
      f0 = *(const float4*)p0; f1 = *(const float4*)(p0 + 4);
      f2 = *(const float4*)p1; f3 = *(const float4*)(p1 + 4);
      if (k0 + 64 < I_DIM) { __builtin_prefetch(p0 + 32, 0, 1); __builtin_prefetch(p1 + 32, 0, 1); }
    }

    v16bf a0 = load_frag_a(&sA[buf][(wm0 + 0)  * LDA], lane);
    v16bf a1 = load_frag_a(&sA[buf][(wm0 + 16) * LDA], lane);
#pragma unroll
    for (int ns = 0; ns < 4; ++ns) {
      v16bf b = load_frag_b(&sB[buf][(wn0 + ns * 16) * LDA], lane);
      acc[0][ns] = __builtin_amdgcn_wmma_f32_16x16x32_bf16(false, a0, false, b, (short)0, acc[0][ns], false, false);
      acc[1][ns] = __builtin_amdgcn_wmma_f32_16x16x32_bf16(false, a1, false, b, (short)0, acc[1][ns], false, false);
    }

    if (more) {
      *(v8bf*)&sB[nb][bofs0] = cvt8(f0, f1);
      *(v8bf*)&sB[nb][bofs1] = cvt8(f2, f3);
    }
    buf = nb;
  }

  const int mlane = (lane >> 4) << 3;
  const int ncol = lane & 15;
  for (int ms = 0; ms < 2; ++ms)
    for (int ns = 0; ns < 4; ++ns) {
      v8f c = acc[ms][ns];
      for (int i = 0; i < 8; ++i) {
        int rloc = wm0 + ms * 16 + mlane + i;
        int rowid = sRow[rloc];
        if (rowid >= 0) {
          int t = rowid >> 1;
          float cf = coefs[rowid];
          int col = n0 + wn0 + ns * 16 + ncol;
          atomicAdd(&out[(size_t)t * H_DIM + col], cf * c[i]);  // exactly 2 adds/elem -> deterministic
        }
      }
    }
}

extern "C" void kernel_launch(void* const* d_in, const int* in_sizes, int n_in,
                              void* d_out, int out_size, void* d_ws, size_t ws_size,
                              hipStream_t stream) {
  (void)in_sizes; (void)n_in; (void)out_size; (void)ws_size;
  const float* x   = (const float*)d_in[0];
  const float* gw  = (const float*)d_in[1];
  const float* ws  = (const float*)d_in[2];
  const float* w2s = (const float*)d_in[3];
  float* out = (float*)d_out;

  // scratch layout (all 256-aligned)
  char* p = (char*)d_ws;
  int*    cnt     = (int*)p;                                     // 16 ints
  int*    toklist = (int*)(p + 256);                             // E*T ints   (256 KB)
  float*  coefs   = (float*)(p + 256 + 262144);                  // 2T floats  (32 KB)
  __bf16* xb      = (__bf16*)(p + 295168);                       // T*H bf16   (16 MB)
  __bf16* h       = (__bf16*)(p + 295168 + (size_t)T_DIM * H_DIM * 2); // 2T*I bf16 (32 MB)

  moe_zero<<<(T_DIM * H_DIM / 4 + 255) / 256, 256, 0, stream>>>(out, T_DIM * H_DIM / 4, cnt);
  moe_gate<<<T_DIM, 256, 0, stream>>>(x, gw, coefs, toklist, cnt);
  moe_cvt<<<T_DIM * H_DIM / 4 / 256, 256, 0, stream>>>(x, xb);
  moe_gemm1<<<dim3(E_NUM, T_DIM / 128, I_DIM / 64), 256, 0, stream>>>(xb, ws, h, toklist, cnt);
  moe_gemm2<<<dim3(E_NUM, T_DIM / 128, H_DIM / 128), 256, 0, stream>>>(h, w2s, out, toklist, coefs, cnt);
}
